// LocalInterface_22548578304859
// MI455X (gfx1250) — compile-verified, matrix operand order checked by reading, not run on hardware
//
#include <hip/hip_runtime.h>

// ESIM co-attention for B=32, P=H=D=512 (fp32 in/out), bf16 WMMA GEMM core.
// Pipeline (all on `stream`):
//   1. pack_tr     : px,hx -> bf16 copies + bf16 transposed copies (LDS tile transpose)
//   2. gemm_e_k    : e = px·hx^T           (v_wmma_f32_16x16x32_bf16, f32 out)
//   3. softmax_row : masked softmax over H -> w_b  (bf16)
//   4. softmax_col : masked softmax over P -> w_a^T (bf16, coalesced transposed store)
//   5. gemm_epi_k  : px_hat = w_b·hx     fused with m_p = [px,x̂,px-x̂,px*x̂]
//   6. gemm_epi_k  : hx_hat = w_a^T·px   fused with m_h
// GEMM core: wave tile 32x64 (8 acc), block tile 128x128 (8 waves, 4Mx2N),
// register double-buffered k-loop (no sched_barrier: round-3 showed pinning the
// schedule causes scratch spills; round-2 codegen is spill-free with good clauses).
// Workspace: 128 MiB.

typedef __bf16 bf16;
typedef bf16  v16bf __attribute__((ext_vector_type(16)));
typedef bf16  v8bf  __attribute__((ext_vector_type(8)));
typedef float v8f   __attribute__((ext_vector_type(8)));

#define NN 512          // P = H = D
#define NB 32           // batch

__device__ __forceinline__ bf16 f2bf(float f) {
    union { float f; unsigned u; } c; c.f = f;
    unsigned r = (c.u + 0x7FFFu + ((c.u >> 16) & 1u)) >> 16;   // RNE
    union { unsigned short s; bf16 b; } o; o.s = (unsigned short)r;
    return o.b;
}

// One 16-element bf16 WMMA fragment for this lane: two contiguous 16B chunks.
__device__ __forceinline__ v16bf ldfrag(const bf16* p) {
    v8bf lo = *(const v8bf*)p;
    v8bf hi = *(const v8bf*)(p + 16);
    return __builtin_shufflevector(lo, hi, 0,1,2,3,4,5,6,7,8,9,10,11,12,13,14,15);
}

// Load one k-step's fragment set: 2 A-frags (rows +0,+16) and 4 B-frags (cols +0..+48).
__device__ __forceinline__ void ld_set(const bf16* __restrict__ Ab,
                                       const bf16* __restrict__ Bb,
                                       int k0, v16bf* fa, v16bf* fb) {
    fa[0] = ldfrag(Ab + k0);
    fa[1] = ldfrag(Ab + 16 * NN + k0);
    fb[0] = ldfrag(Bb + k0);
    fb[1] = ldfrag(Bb + 16 * NN + k0);
    fb[2] = ldfrag(Bb + 32 * NN + k0);
    fb[3] = ldfrag(Bb + 48 * NN + k0);
}

#define WMMA_STEP(FA, FB, ACC)                                                        \
    _Pragma("unroll") for (int i = 0; i < 2; ++i)                                     \
    _Pragma("unroll") for (int j = 0; j < 4; ++j)                                     \
        ACC[i][j] = __builtin_amdgcn_wmma_f32_16x16x32_bf16(                          \
            false, FA[i], false, FB[j], (short)0, ACC[i][j], false, false);

// ---------------------------------------------------------------- pack + transpose
__global__ void pack_tr(const float* __restrict__ src,
                        bf16* __restrict__ dstN, bf16* __restrict__ dstT) {
    __shared__ float tile[32][33];
    int blk = blockIdx.x;
    int b = blk >> 8;                 // 256 tiles (16x16) per batch matrix
    int t = blk & 255;
    int p0 = (t >> 4) << 5;
    int d0 = (t & 15) << 5;
    int tx = threadIdx.x & 31;
    int ty = threadIdx.x >> 5;        // 0..7
    const float* s  = src  + ((size_t)b << 18);
    bf16* dn = dstN + ((size_t)b << 18);
    bf16* dt = dstT + ((size_t)b << 18);
#pragma unroll
    for (int y = 0; y < 4; ++y) {
        int pl = ty + (y << 3);
        float v = s[(size_t)(p0 + pl) * NN + d0 + tx];
        dn[(size_t)(p0 + pl) * NN + d0 + tx] = f2bf(v);
        tile[pl][tx] = v;
    }
    __syncthreads();
#pragma unroll
    for (int y = 0; y < 4; ++y) {
        int dl = ty + (y << 3);
        dt[(size_t)(d0 + dl) * NN + p0 + tx] = f2bf(tile[tx][dl]);
    }
}

// ---------------------------------------------------------------- e = A · Bt^T (NT GEMM)
// Block tile 128x128 (8 waves = 4M x 2N), wave tile 32x64, double-buffered k-loop.
__global__ void gemm_e_k(const bf16* __restrict__ A, const bf16* __restrict__ Bt,
                         float* __restrict__ E) {
    int lane = threadIdx.x & 31;
    int w    = threadIdx.x >> 5;       // 0..7
    int l16  = lane & 15;
    int hi   = lane >> 4;
    int b  = blockIdx.x >> 4;          // 16 blocks per batch
    int t  = blockIdx.x & 15;
    int Mb = ((t >> 2) << 7) + ((w >> 1) << 5);   // mB*128 + mW*32
    int Nb = ((t & 3)  << 7) + ((w & 1) << 6);    // nB*128 + nW*64
    const bf16* Ab = A  + ((size_t)b << 18) + (size_t)(Mb + l16) * NN + hi * 8;
    const bf16* Bb = Bt + ((size_t)b << 18) + (size_t)(Nb + l16) * NN + hi * 8;

    v8f acc[2][4];
#pragma unroll
    for (int i = 0; i < 2; ++i)
#pragma unroll
        for (int j = 0; j < 4; ++j) acc[i][j] = (v8f){};

    v16bf fa[2][2], fb[2][4];
    ld_set(Ab, Bb, 0, fa[0], fb[0]);
#pragma unroll
    for (int k0 = 0; k0 < NN; k0 += 32) {
        int cur = (k0 >> 5) & 1, nxt = cur ^ 1;
        if (k0 + 32 < NN) ld_set(Ab, Bb, k0 + 32, fa[nxt], fb[nxt]);
        WMMA_STEP(fa[cur], fb[cur], acc)
    }

    float* Eb = E + ((size_t)b << 18);
#pragma unroll
    for (int i = 0; i < 2; ++i) {
        int row0 = Mb + i * 16 + hi * 8;
#pragma unroll
        for (int rr = 0; rr < 8; ++rr) {
            float* er = Eb + (size_t)(row0 + rr) * NN + Nb + l16;
            er[0]  = acc[i][0][rr];
            er[16] = acc[i][1][rr];
            er[32] = acc[i][2][rr];
            er[48] = acc[i][3][rr];
        }
    }
}

// ---------------------------------------------------------------- masked row softmax (over H)
__global__ void softmax_row_k(const float* __restrict__ E,
                              const unsigned char* __restrict__ hmask,
                              bf16* __restrict__ W) {
    int wid  = (int)((blockIdx.x * blockDim.x + threadIdx.x) >> 5);  // (b,p)
    int lane = threadIdx.x & 31;
    int b = wid >> 9;
    const float* row = E + ((size_t)wid << 9);
    const unsigned char* mk = hmask + ((size_t)b << 9);
    float v[16];
    float m = -__builtin_inff();
#pragma unroll
    for (int i = 0; i < 16; ++i) {
        int h = lane + (i << 5);
        float x = mk[h] ? -__builtin_inff() : row[h];
        v[i] = x; m = fmaxf(m, x);
    }
#pragma unroll
    for (int s = 16; s > 0; s >>= 1) m = fmaxf(m, __shfl_xor(m, s, 32));
    float sum = 0.f;
#pragma unroll
    for (int i = 0; i < 16; ++i) { float tv = __expf(v[i] - m); v[i] = tv; sum += tv; }
#pragma unroll
    for (int s = 16; s > 0; s >>= 1) sum += __shfl_xor(sum, s, 32);
    float inv = 1.0f / sum;
    bf16* wp = W + ((size_t)wid << 9);
#pragma unroll
    for (int i = 0; i < 16; ++i) wp[lane + (i << 5)] = f2bf(v[i] * inv);
}

// ---------------------------------------------------------------- masked col softmax (over P)
// Writes w_a^T[b][h][p] row-major -> coalesced store + contiguous A-frags later.
__global__ void softmax_col_k(const float* __restrict__ E,
                              const unsigned char* __restrict__ pmask,
                              bf16* __restrict__ Wt) {
    int wid  = (int)((blockIdx.x * blockDim.x + threadIdx.x) >> 5);  // (b,h)
    int lane = threadIdx.x & 31;
    int b = wid >> 9;
    int h = wid & 511;
    const float* col = E + ((size_t)b << 18) + h;
    const unsigned char* mk = pmask + ((size_t)b << 9);
    float v[16];
    float m = -__builtin_inff();
#pragma unroll
    for (int i = 0; i < 16; ++i) {
        int p = lane + (i << 5);
        float x = mk[p] ? -__builtin_inff() : col[(size_t)p << 9];
        v[i] = x; m = fmaxf(m, x);
    }
#pragma unroll
    for (int s = 16; s > 0; s >>= 1) m = fmaxf(m, __shfl_xor(m, s, 32));
    float sum = 0.f;
#pragma unroll
    for (int i = 0; i < 16; ++i) { float tv = __expf(v[i] - m); v[i] = tv; sum += tv; }
#pragma unroll
    for (int s = 16; s > 0; s >>= 1) sum += __shfl_xor(sum, s, 32);
    float inv = 1.0f / sum;
    bf16* wp = Wt + ((size_t)wid << 9);
#pragma unroll
    for (int i = 0; i < 16; ++i) wp[lane + (i << 5)] = f2bf(v[i] * inv);
}

// ---------------------------------------------------------------- NT GEMM + concat epilogue
// C[row,col] = sum_k A[row,k]*Bt[col,k];  out[row] = [x, c, x-c, x*c] slabs of 512.
__global__ void gemm_epi_k(const bf16* __restrict__ A, const bf16* __restrict__ Bt,
                           const float* __restrict__ X, float* __restrict__ out) {
    int lane = threadIdx.x & 31;
    int w    = threadIdx.x >> 5;
    int l16  = lane & 15;
    int hi   = lane >> 4;
    int b  = blockIdx.x >> 4;
    int t  = blockIdx.x & 15;
    int Mb = ((t >> 2) << 7) + ((w >> 1) << 5);
    int Nb = ((t & 3)  << 7) + ((w & 1) << 6);
    const bf16* Ab = A  + ((size_t)b << 18) + (size_t)(Mb + l16) * NN + hi * 8;
    const bf16* Bb = Bt + ((size_t)b << 18) + (size_t)(Nb + l16) * NN + hi * 8;

    v8f acc[2][4];
#pragma unroll
    for (int i = 0; i < 2; ++i)
#pragma unroll
        for (int j = 0; j < 4; ++j) acc[i][j] = (v8f){};

    v16bf fa[2][2], fb[2][4];
    ld_set(Ab, Bb, 0, fa[0], fb[0]);
#pragma unroll
    for (int k0 = 0; k0 < NN; k0 += 32) {
        int cur = (k0 >> 5) & 1, nxt = cur ^ 1;
        if (k0 + 32 < NN) ld_set(Ab, Bb, k0 + 32, fa[nxt], fb[nxt]);
        WMMA_STEP(fa[cur], fb[cur], acc)
    }

    const float* Xb = X + ((size_t)b << 18);
    float* Ob = out + ((size_t)b << 20);          // b * 512 * 2048
#pragma unroll
    for (int i = 0; i < 2; ++i) {
        int row0 = Mb + i * 16 + hi * 8;
#pragma unroll
        for (int rr = 0; rr < 8; ++rr) {
            int row = row0 + rr;
#pragma unroll
            for (int j = 0; j < 4; ++j) {
                int col = Nb + j * 16 + l16;
                float x = Xb[(size_t)row * NN + col];
                float c = acc[i][j][rr];
                float* o = Ob + (size_t)row * 2048 + col;
                o[0]    = x;
                o[512]  = c;
                o[1024] = x - c;
                o[1536] = x * c;
            }
        }
    }
}

// ---------------------------------------------------------------- launcher
extern "C" void kernel_launch(void* const* d_in, const int* in_sizes, int n_in,
                              void* d_out, int out_size, void* d_ws, size_t ws_size,
                              hipStream_t stream) {
    (void)in_sizes; (void)n_in; (void)out_size; (void)ws_size;
    const float*         px     = (const float*)d_in[0];
    const float*         hx     = (const float*)d_in[1];
    const unsigned char* p_mask = (const unsigned char*)d_in[2];  // jax bool = 1 byte
    const unsigned char* h_mask = (const unsigned char*)d_in[3];
    float* out = (float*)d_out;

    char* ws = (char*)d_ws;
    const size_t MB16 = (size_t)16 << 20;
    bf16*  px_bf = (bf16*)(ws);
    bf16*  hx_bf = (bf16*)(ws + 1 * MB16);
    bf16*  pxT   = (bf16*)(ws + 2 * MB16);
    bf16*  hxT   = (bf16*)(ws + 3 * MB16);
    float* E     = (float*)(ws + 4 * MB16);       // 32 MiB
    bf16*  wb    = (bf16*)(ws + 6 * MB16);
    bf16*  waT   = (bf16*)(ws + 7 * MB16);        // total 128 MiB

    dim3 blk(256);
    // 1) bf16 pack + transpose (8192 = 32 batches * 16x16 tiles)
    pack_tr<<<8192, blk, 0, stream>>>(px, px_bf, pxT);
    pack_tr<<<8192, blk, 0, stream>>>(hx, hx_bf, hxT);
    // 2) e = px · hx^T   (512 blocks: 32 batches * 4x4 block tiles of 128x128)
    gemm_e_k<<<512, blk, 0, stream>>>(px_bf, hx_bf, E);
    // 3) softmax over H (16384 waves -> 2048 blocks)
    softmax_row_k<<<2048, blk, 0, stream>>>(E, h_mask, wb);
    // 4) softmax over P -> w_a^T
    softmax_col_k<<<2048, blk, 0, stream>>>(E, p_mask, waT);
    // 5) m_p: px_hat = w_b · hx  (B as hxT rows), epilogue with px
    gemm_epi_k<<<512, blk, 0, stream>>>(wb, hxT, px, out);
    // 6) m_h: hx_hat = w_a^T · px (B as pxT rows), epilogue with hx
    gemm_epi_k<<<512, blk, 0, stream>>>(waT, pxT, hx, out + (size_t)NB * NN * 2048);
}
